// RelBlockMultiHeadedAttention_14336600834262
// MI455X (gfx1250) — matrix operand order repaired and checked by reading, not compiled, not run on hardware
//
#include <hip/hip_runtime.h>
#include <hip/hip_bf16.h>
#include <math.h>

typedef __attribute__((ext_vector_type(16))) _Float16 v16h;
typedef __attribute__((ext_vector_type(8)))  float    v8f;
typedef __attribute__((ext_vector_type(4)))  unsigned int u32x4;
typedef __attribute__((ext_vector_type(8)))  int      i32x8;
typedef __attribute__((ext_vector_type(4)))  int      i32x4;
typedef _Float16 half_t;

#define Hh   8
#define DK   64
#define Dd   512
#define BLEN 256
#define Tt   2048
#define Bb   8
#define NBLK 8
#define K2   512

// ---------------------------------------------------------------------------
// WMMA helpers (CDNA5 16x16x32 f16 -> f32), layouts per ISA 7.12.2
// ---------------------------------------------------------------------------
__device__ inline v8f wmma16(v16h a, v16h b, v8f c) {
  return __builtin_amdgcn_wmma_f32_16x16x32_f16(
      false, a, false, b, (short)0, c, false, false);
}

// A fragment: 16x32 (MxK) f16, src row-major [m][k], row stride ld (halves)
__device__ inline v16h load_a(const half_t* src, int ld, int m0, int k0, int lane) {
  int m  = m0 + (lane & 15);
  int hi = (lane >> 4) & 1;
  const half_t* p = src + (size_t)m * ld + k0 + hi * 8;
  v16h a;
#pragma unroll
  for (int i = 0; i < 8; ++i) a[i] = p[i];
#pragma unroll
  for (int i = 0; i < 8; ++i) a[8 + i] = p[16 + i];
  return a;
}

// A fragment with per-K bias add (q + pos_bias)
__device__ inline v16h load_a_bias(const half_t* src, int ld, int m0, int k0,
                                   int lane, const half_t* bias) {
  int m  = m0 + (lane & 15);
  int hi = (lane >> 4) & 1;
  const half_t* p  = src + (size_t)m * ld + k0 + hi * 8;
  const half_t* bp = bias + k0 + hi * 8;
  v16h a;
#pragma unroll
  for (int i = 0; i < 8; ++i) a[i] = (half_t)(p[i] + bp[i]);
#pragma unroll
  for (int i = 0; i < 8; ++i) a[8 + i] = (half_t)(p[16 + i] + bp[16 + i]);
  return a;
}

// A fragment from f32 source (softmax probs), converting to f16
__device__ inline v16h load_a_f32(const float* src, int ld, int m0, int k0, int lane) {
  int m  = m0 + (lane & 15);
  int hi = (lane >> 4) & 1;
  const float* p = src + (size_t)m * ld + k0 + hi * 8;
  v16h a;
#pragma unroll
  for (int i = 0; i < 8; ++i) a[i] = (half_t)p[i];
#pragma unroll
  for (int i = 0; i < 8; ++i) a[8 + i] = (half_t)p[16 + i];
  return a;
}

// B fragment: 32x16 (KxN). src is "N-major": src[n][k], row stride ld.
__device__ inline v16h load_b(const half_t* src, int ld, int k0, int n0, int lane) {
  int n  = n0 + (lane & 15);
  int kk = k0 + ((lane >> 4) & 1) * 8;
  const half_t* p = src + (size_t)n * ld + kk;
  v16h b;
#pragma unroll
  for (int i = 0; i < 8; ++i) b[i] = p[i];
#pragma unroll
  for (int i = 0; i < 8; ++i) b[8 + i] = p[16 + i];
  return b;
}

// C store: lane n = n0+(l&15); VGPR j -> row m0 + j + (l>=16 ? 8 : 0)
__device__ inline void store_c(float* dst, int ld, int m0, int n0, int lane, v8f c) {
  int n  = n0 + (lane & 15);
  int mb = m0 + ((lane >> 4) & 1) * 8;
#pragma unroll
  for (int j = 0; j < 8; ++j) dst[(size_t)(mb + j) * ld + n] = c[j];
}

// ---------------------------------------------------------------------------
// Tensor Data Mover: 2-D tile (rows x 64 f16 elems, row stride in elems)
// from global to LDS. Descriptor per ISA cdna5 ch.8 (D# groups 0/1).
// ---------------------------------------------------------------------------
#if __has_builtin(__builtin_amdgcn_tensor_load_to_lds)
#define USE_TDM 1
#endif

__device__ inline unsigned lds_off32(const void* p) {
  // Flat LDS-aperture addresses carry the LDS byte offset in the low 32 bits.
  return (unsigned)(unsigned long long)(size_t)p;
}

__device__ inline void wait_tensorcnt0() {
#if __has_builtin(__builtin_amdgcn_s_wait_tensorcnt)
  __builtin_amdgcn_s_wait_tensorcnt(0);
#elif defined(__AMDGCN__)
  asm volatile("s_wait_tensorcnt 0x0" ::: "memory");
#endif
}

#if defined(USE_TDM)
__device__ inline void tdm_load_2d(const half_t* gaddr, unsigned lds_bytes,
                                   unsigned rows, unsigned row_stride_elems) {
  unsigned long long ga = (unsigned long long)(size_t)gaddr;
  u32x4 g0;
  g0[0] = 1u;                                        // count=1 (valid), user mode
  g0[1] = lds_bytes;                                 // lds_addr
  g0[2] = (unsigned)(ga & 0xFFFFFFFFu);              // global_addr lo
  g0[3] = (unsigned)((ga >> 32) & 0x01FFFFFFu) | (2u << 30);  // ga hi | type=2
  i32x8 g1;
  g1[0] = (int)(1u << 16);                           // data_size=1 (2 bytes)
  g1[1] = (int)(64u << 16);                          // tensor_dim0 = 64 (lo16)
  g1[2] = (int)((rows & 0xFFFFu) << 16);             // dim0 hi=0 | tensor_dim1 lo16
  g1[3] = (int)((rows >> 16) | (64u << 16));         // tensor_dim1 hi | tile_dim0=64
  g1[4] = (int)(rows & 0xFFFFu);                     // tile_dim1=rows, tile_dim2=0
  g1[5] = (int)row_stride_elems;                     // tensor_dim0_stride lo32
  g1[6] = 0;                                         // stride hi | dim1_stride lo
  g1[7] = 0;
  i32x4 g2 = {0, 0, 0, 0};
  i32x4 g3 = {0, 0, 0, 0};
#if __clang_major__ >= 23
  i32x8 g4 = {0, 0, 0, 0, 0, 0, 0, 0};
  __builtin_amdgcn_tensor_load_to_lds(g0, g1, g2, g3, g4, 0);
#else
  __builtin_amdgcn_tensor_load_to_lds(g0, g1, g2, g3, 0);
#endif
}
#endif

// ---------------------------------------------------------------------------
// Generic GEMM: C[M,512] = A[M,512] @ W[512,512] + bias
// grid (M/64, 8), block 128 (4 waves, 16 rows x 64 cols each), K-tile 64
// ---------------------------------------------------------------------------
template <int A_HALF, int OUT_FLOAT>
__global__ __launch_bounds__(128) void gemm512_kernel(
    const void* __restrict__ Aptr, const float* __restrict__ W,
    const float* __restrict__ bias, void* __restrict__ outp) {
  __shared__ half_t sA[64 * 64];    // [m][k]
  __shared__ half_t sBt[64 * 64];   // [n][k]  (W tile transposed)
  const int tid = threadIdx.x, lane = tid & 31, wave = tid >> 5;
  const int mblk = blockIdx.x * 64, nblk = blockIdx.y * 64;
  const float*  Af = (const float*)Aptr;
  const half_t* Ah = (const half_t*)Aptr;
  v8f acc[4] = {{0}, {0}, {0}, {0}};

  for (int k0 = 0; k0 < 512; k0 += 64) {
    __syncthreads();
    for (int idx = tid; idx < 64 * 64; idx += 128) {
      int r = idx >> 6, c = idx & 63;
      size_t g = (size_t)(mblk + r) * 512 + k0 + c;
      sA[idx] = A_HALF ? Ah[g] : (half_t)Af[g];
    }
    for (int idx = tid; idx < 64 * 64; idx += 128) {
      int n = idx >> 6, k = idx & 63;
      sBt[idx] = (half_t)W[(size_t)(k0 + k) * 512 + nblk + n];
    }
    __syncthreads();
    v16h a0 = load_a(sA, 64, wave * 16, 0,  lane);
    v16h a1 = load_a(sA, 64, wave * 16, 32, lane);
    v16h b00 = load_b(sBt, 64, 0,  0,  lane);
    v16h b01 = load_b(sBt, 64, 0,  16, lane);
    v16h b02 = load_b(sBt, 64, 0,  32, lane);
    v16h b03 = load_b(sBt, 64, 0,  48, lane);
    acc[0] = wmma16(a0, b00, acc[0]);
    acc[1] = wmma16(a0, b01, acc[1]);
    acc[2] = wmma16(a0, b02, acc[2]);
    acc[3] = wmma16(a0, b03, acc[3]);
    v16h b10 = load_b(sBt, 64, 32, 0,  lane);
    v16h b11 = load_b(sBt, 64, 32, 16, lane);
    v16h b12 = load_b(sBt, 64, 32, 32, lane);
    v16h b13 = load_b(sBt, 64, 32, 48, lane);
    acc[0] = wmma16(a1, b10, acc[0]);
    acc[1] = wmma16(a1, b11, acc[1]);
    acc[2] = wmma16(a1, b12, acc[2]);
    acc[3] = wmma16(a1, b13, acc[3]);
  }

  int nl = lane & 15, mb = ((lane >> 4) & 1) * 8;
#pragma unroll
  for (int nt = 0; nt < 4; ++nt) {
    int n = nblk + nt * 16 + nl;
    float bv = bias[n];
#pragma unroll
    for (int j = 0; j < 8; ++j) {
      int m = mblk + wave * 16 + mb + j;
      float v = acc[nt][j] + bv;
      if (OUT_FLOAT) ((float*)outp)[(size_t)m * 512 + n] = v;
      else           ((half_t*)outp)[(size_t)m * 512 + n] = (half_t)v;
    }
  }
}

// ---------------------------------------------------------------------------
// Sinusoidal positional encoding, PE[512][512] f32
// ---------------------------------------------------------------------------
__global__ void pe_kernel(float* __restrict__ PE) {
  int idx = blockIdx.x * blockDim.x + threadIdx.x;
  if (idx >= 512 * 512) return;
  int pos = idx >> 9, c = idx & 511;
  int i = c >> 1;
  float div = __expf((float)(2 * i) * (-9.210340371976184f / 512.0f));
  float ang = (float)pos * div;
  PE[idx] = (c & 1) ? __cosf(ang) : __sinf(ang);
}

// ---------------------------------------------------------------------------
// Block relative attention.
// grid (8 mchunks, 8 heads, 64 block-batches), block 64 threads (2 waves).
// ---------------------------------------------------------------------------
__global__ __launch_bounds__(64) void relattn_kernel(
    const half_t* __restrict__ Qf, const half_t* __restrict__ Kf,
    const half_t* __restrict__ Vf, const half_t* __restrict__ Rf,
    const float* __restrict__ pos_u, const float* __restrict__ pos_v,
    const int* __restrict__ mask, half_t* __restrict__ Xf) {
  __shared__ half_t sQ[32 * 64];         // q rows, raw
  __shared__ half_t sKRV[512 * 64];      // K[t][d] -> r[t][d] -> V^T[d][t]
  __shared__ float  sS[32 * 512];        // scores / probs
  __shared__ half_t sBias[2 * 64];       // u, v for this head
  __shared__ unsigned char sValid[512];

  const int mchunk = blockIdx.x;         // 0..7  (32 q-rows each)
  const int h      = blockIdx.y;         // 0..7
  const int bn     = blockIdx.z;         // 0..63
  const int b  = bn / NBLK;
  const int jb = bn % NBLK;
  const int tid = threadIdx.x, lane = tid & 31, wave = tid >> 5;
  const int qrow0 = mchunk * 32;

  // ---- stage Q rows, biases, key-validity ----
  for (int idx = tid; idx < 32 * 64; idx += 64) {
    int r = idx >> 6, d = idx & 63;
    int gq = jb * BLEN + qrow0 + r;
    sQ[idx] = Qf[(size_t)(b * Tt + gq) * Dd + h * DK + d];
  }
  if (tid < 64) {
    sBias[tid]      = (half_t)pos_u[h * DK + tid];
    sBias[64 + tid] = (half_t)pos_v[h * DK + tid];
  }
  for (int t = tid; t < 512; t += 64) {
    int pos = jb * BLEN + t - BLEN;
    int ok = (pos >= 0 && pos < Tt) ? mask[b * Tt + pos] : 0;
    sValid[t] = (unsigned char)(ok != 0);
  }

  // ---- K tile: TDM 2-D tile load (zero-fill leading rows for block 0) ----
#if defined(USE_TDM)
  if (jb == 0) {
    for (int idx = tid; idx < 256 * 64; idx += 64) sKRV[idx] = (half_t)0.0f;
    if (wave == 0) {
      tdm_load_2d(Kf + (size_t)(b * Tt) * Dd + h * DK,
                  lds_off32(sKRV) + 256 * 64 * (unsigned)sizeof(half_t),
                  256, 512);
      wait_tensorcnt0();
    }
  } else {
    if (wave == 0) {
      tdm_load_2d(Kf + (size_t)(b * Tt + jb * BLEN - BLEN) * Dd + h * DK,
                  lds_off32(sKRV), 512, 512);
      wait_tensorcnt0();
    }
  }
#else
  for (int idx = tid; idx < 512 * 64; idx += 64) {
    int t = idx >> 6, d = idx & 63;
    int pos = jb * BLEN + t - BLEN;
    half_t val = (half_t)0.0f;
    if (pos >= 0 && pos < Tt)
      val = Kf[(size_t)(b * Tt + pos) * Dd + h * DK + d];
    sKRV[idx] = val;
  }
#endif
  __syncthreads();

  // ---- phase 1: S1 = (q + u) @ K^T -> sS ----
  {
    int m0 = wave * 16;
    v16h a0 = load_a_bias(sQ, 64, m0, 0,  lane, sBias);
    v16h a1 = load_a_bias(sQ, 64, m0, 32, lane, sBias);
    for (int n0 = 0; n0 < 512; n0 += 32) {
      v16h bA0 = load_b(sKRV, 64, 0,  n0,      lane);
      v16h bA1 = load_b(sKRV, 64, 32, n0,      lane);
      v16h bB0 = load_b(sKRV, 64, 0,  n0 + 16, lane);
      v16h bB1 = load_b(sKRV, 64, 32, n0 + 16, lane);
      v8f c0 = {0};
      c0 = wmma16(a0, bA0, c0);
      c0 = wmma16(a1, bA1, c0);
      v8f c1 = {0};
      c1 = wmma16(a0, bB0, c1);
      c1 = wmma16(a1, bB1, c1);
      store_c(sS, 512, m0, n0,      lane, c0);
      store_c(sS, 512, m0, n0 + 16, lane, c1);
    }
  }
  __syncthreads();

  // ---- phase 2: load r tile via TDM (overwrites K) ----
#if defined(USE_TDM)
  if (wave == 0) {
    tdm_load_2d(Rf + h * DK, lds_off32(sKRV), 512, 512);
    wait_tensorcnt0();
  }
#else
  for (int idx = tid; idx < 512 * 64; idx += 64) {
    int t = idx >> 6, d = idx & 63;
    sKRV[idx] = Rf[(size_t)t * Dd + h * DK + d];
  }
#endif
  __syncthreads();

  // ---- phase 3: S2 = (q + v) @ r^T, scatter-add with rel-shift ----
  // shifted[q,k] takes S2[q,j]: j = k-q+255 (if <512) else j = 767-k+q.
  // Inverse (injective): S2[q,j] -> k1 = j+q-255 ; and k2 = q+767-j if j<=510.
  {
    int m0 = wave * 16;
    v16h a0 = load_a_bias(sQ, 64, m0, 0,  lane, sBias + 64);
    v16h a1 = load_a_bias(sQ, 64, m0, 32, lane, sBias + 64);
    int nl = lane & 15, mb = ((lane >> 4) & 1) * 8;
    for (int n0 = 0; n0 < 512; n0 += 16) {
      v16h b0 = load_b(sKRV, 64, 0,  n0, lane);
      v16h b1 = load_b(sKRV, 64, 32, n0, lane);
      v8f c = {0};
      c = wmma16(a0, b0, c);
      c = wmma16(a1, b1, c);
      int jcol = n0 + nl;
#pragma unroll
      for (int j = 0; j < 8; ++j) {
        int mloc = m0 + mb + j;            // local row (this wave's rows only)
        int q    = qrow0 + mloc;           // absolute q in block, 0..255
        float val = c[j];
        int k1 = jcol + q - 255;
        if (k1 >= 0 && k1 < 512) sS[(size_t)mloc * 512 + k1] += val;
        int k2 = q + 767 - jcol;
        if (jcol <= 510 && k2 >= 0 && k2 < 512) sS[(size_t)mloc * 512 + k2] += val;
      }
    }
  }
  __syncthreads();

  // ---- phase 4: masked softmax in place (2 lanes per row) ----
  {
    int row   = wave * 16 + (lane & 15);
    int half_ = lane >> 4;
    float* srow = sS + (size_t)row * 512;
    int c0 = half_ * 256, c1 = c0 + 256;
    float mx = -3.4e38f;
    for (int k = c0; k < c1; ++k) {
      float s = sValid[k] ? (srow[k] * 0.125f) : -3.4e38f;
      srow[k] = s;
      mx = fmaxf(mx, s);
    }
    mx = fmaxf(mx, __shfl_xor(mx, 16, 32));
    float sum = 0.0f;
    for (int k = c0; k < c1; ++k) {
      float e = __expf(srow[k] - mx);
      srow[k] = e;
      sum += e;
    }
    sum += __shfl_xor(sum, 16, 32);
    float inv = 1.0f / sum;
    for (int k = c0; k < c1; ++k)
      srow[k] = sValid[k] ? srow[k] * inv : 0.0f;
  }
  __syncthreads();

  // ---- phase 5: load V transposed (V^T[d][t]) over same LDS ----
  for (int idx = tid; idx < 512 * 64; idx += 64) {
    int t = idx >> 6, d = idx & 63;
    int pos = jb * BLEN + t - BLEN;
    half_t val = (half_t)0.0f;
    if (pos >= 0 && pos < Tt)
      val = Vf[(size_t)(b * Tt + pos) * Dd + h * DK + d];
    sKRV[(size_t)d * 512 + t] = val;
  }
  __syncthreads();

  // ---- phase 6: x = P @ V  (16x64 per wave, K=512) ----
  {
    int m0 = wave * 16;
    v8f acc[4] = {{0}, {0}, {0}, {0}};
    for (int k0 = 0; k0 < 512; k0 += 32) {
      v16h a  = load_a_f32(sS, 512, m0, k0, lane);
      v16h b0 = load_b(sKRV, 512, k0, 0,  lane);
      v16h b1 = load_b(sKRV, 512, k0, 16, lane);
      v16h b2 = load_b(sKRV, 512, k0, 32, lane);
      v16h b3 = load_b(sKRV, 512, k0, 48, lane);
      acc[0] = wmma16(a, b0, acc[0]);
      acc[1] = wmma16(a, b1, acc[1]);
      acc[2] = wmma16(a, b2, acc[2]);
      acc[3] = wmma16(a, b3, acc[3]);
    }
    int nl = lane & 15, mb = ((lane >> 4) & 1) * 8;
#pragma unroll
    for (int nt = 0; nt < 4; ++nt) {
      int d = nt * 16 + nl;
#pragma unroll
      for (int j = 0; j < 8; ++j) {
        int mloc = m0 + mb + j;
        int gq = jb * BLEN + qrow0 + mloc;
        Xf[(size_t)(b * Tt + gq) * Dd + h * DK + d] = (half_t)acc[nt][j];
      }
    }
  }
}

// ---------------------------------------------------------------------------
extern "C" void kernel_launch(void* const* d_in, const int* in_sizes, int n_in,
                              void* d_out, int out_size, void* d_ws, size_t ws_size,
                              hipStream_t stream) {
  const float* query = (const float*)d_in[0];
  const float* key   = (const float*)d_in[1];
  const float* value = (const float*)d_in[2];
  const int*   mask  = (const int*)d_in[3];
  const float* w_q = (const float*)d_in[4];
  const float* b_q = (const float*)d_in[5];
  const float* w_k = (const float*)d_in[6];
  const float* b_k = (const float*)d_in[7];
  const float* w_v = (const float*)d_in[8];
  const float* b_v = (const float*)d_in[9];
  const float* w_r = (const float*)d_in[10];
  const float* b_r = (const float*)d_in[11];
  const float* w_o = (const float*)d_in[12];
  const float* b_o = (const float*)d_in[13];
  const float* pos_u = (const float*)d_in[14];
  const float* pos_v = (const float*)d_in[15];

  const size_t MT = (size_t)Bb * Tt;        // 16384 rows
  char* ws = (char*)d_ws;
  size_t off = 0;
  half_t* Qf = (half_t*)(ws + off); off += MT * Dd * sizeof(half_t);   // 16 MB
  half_t* Kf = (half_t*)(ws + off); off += MT * Dd * sizeof(half_t);
  half_t* Vf = (half_t*)(ws + off); off += MT * Dd * sizeof(half_t);
  half_t* Xf = (half_t*)(ws + off); off += MT * Dd * sizeof(half_t);
  half_t* Rf = (half_t*)(ws + off); off += (size_t)K2 * Dd * sizeof(half_t);
  float*  PE = (float*)(ws + off);  off += (size_t)K2 * Dd * sizeof(float);

  dim3 gproj(MT / 64, Dd / 64);   // 256 x 8
  gemm512_kernel<0, 0><<<gproj, 128, 0, stream>>>(query, w_q, b_q, Qf);
  gemm512_kernel<0, 0><<<gproj, 128, 0, stream>>>(key,   w_k, b_k, Kf);
  gemm512_kernel<0, 0><<<gproj, 128, 0, stream>>>(value, w_v, b_v, Vf);

  pe_kernel<<<(512 * 512 + 255) / 256, 256, 0, stream>>>(PE);
  dim3 gr(K2 / 64, Dd / 64);      // 8 x 8
  gemm512_kernel<0, 0><<<gr, 128, 0, stream>>>(PE, w_r, b_r, Rf);

  dim3 gattn(8, Hh, Bb * NBLK);   // (mchunk, head, block-batch)
  relattn_kernel<<<gattn, 64, 0, stream>>>(Qf, Kf, Vf, Rf, pos_u, pos_v, mask, Xf);

  gemm512_kernel<1, 1><<<gproj, 128, 0, stream>>>(Xf, w_o, b_o, d_out);
}
